// OrthoLinear_26792005993020
// MI455X (gfx1250) — matrix operand order repaired and checked by reading, not compiled, not run on hardware
//
#include <hip/hip_runtime.h>

// ---------------------------------------------------------------------------
// OrthoLinear: y = x @ (dequant_int4(W) + CSR)^T
//   x:   [4, 2048, 4096] fp32 -> M = 8192
//   W:   [4096, 4096] int4+scale+CSR -> materialized as f16 in d_ws (33.5 MB)
//   out: [4, 2048, 4096] fp32
// GEMM: M=8192, N=4096, K=4096, ~275 GFLOP -> matrix-core bound on MI455X.
// Block tile 128x256, BK=32, 8 wave32 (2x4), wave tile 64x64 -> 16 WMMAs/step.
// A (x) staged via regs with fp32->f16 cvt; B (W16) staged with
// GLOBAL_LOAD_ASYNC_TO_LDS_B128 (ASYNCcnt) - no register round-trip.
// ---------------------------------------------------------------------------

#define M_ROWS 8192
#define N_OUT  4096
#define K_IN   4096
#define NNZ_PER_ROW 204

#define BM 128
#define BN 256
#define BK 32
#define K_TILES (K_IN / BK)   // 128
#define LDSTRIDE 40           // 32 halves + 8 pad (80B rows -> spread banks)

typedef __attribute__((ext_vector_type(16))) _Float16 v16h;
typedef __attribute__((ext_vector_type(8)))  _Float16 v8h;
typedef __attribute__((ext_vector_type(8)))  float    v8f;
typedef __attribute__((ext_vector_type(4)))  float    v4f;

// -------- CDNA5 async global->LDS copy helpers (inline asm, ISA §10/15.18) --
__device__ __forceinline__ void async_b128(unsigned lds_off, const void* gptr) {
    asm volatile("global_load_async_to_lds_b128 %0, %1, off"
                 :: "v"(lds_off), "v"(gptr) : "memory");
}
__device__ __forceinline__ void wait_async0() {
    asm volatile("s_wait_asynccnt 0" ::: "memory");
}
__device__ __forceinline__ unsigned lds_addr(const void* p) {
    // flat LDS pointers carry the LDS byte offset in the low 32 bits
    return (unsigned)(unsigned long long)p;
}

// ---------------------------------------------------------------------------
// Pass 1: dequantize one W row per block into f16 (+ CSR via LDS f32 atomics)
// ---------------------------------------------------------------------------
__global__ __launch_bounds__(256)
void ortho_dequant_kernel(const int* __restrict__ base_packed,
                          const float* __restrict__ scales,
                          const _Float16* __restrict__ ortho_vals,
                          const int* __restrict__ ortho_cols,
                          _Float16* __restrict__ W16)
{
    __shared__ float row[K_IN];                  // 16 KB
    const int o = blockIdx.x;
    const float s = scales[o];
    const int* rp = base_packed + (size_t)o * (K_IN / 2);

    for (int j = threadIdx.x; j < K_IN / 2; j += 256) {
        int p = rp[j];
        row[2 * j]     = (float)((p        & 0xF) - 8) * s;
        row[2 * j + 1] = (float)(((p >> 4) & 0xF) - 8) * s;
    }
    __syncthreads();

    const int base = o * NNZ_PER_ROW;
    for (int j = threadIdx.x; j < NNZ_PER_ROW; j += 256) {
        atomicAdd(&row[ortho_cols[base + j]], (float)ortho_vals[base + j]); // ALPHA=1
    }
    __syncthreads();

    _Float16* wrow = W16 + (size_t)o * K_IN;
    for (int j = threadIdx.x; j < K_IN; j += 256) {
        wrow[j] = (_Float16)row[j];
    }
}

// ---------------------------------------------------------------------------
// Pass 2: WMMA f16 GEMM, out[M,N] = x[M,K] * W16[N,K]^T
// ---------------------------------------------------------------------------
__global__ __launch_bounds__(256)
void ortho_gemm_kernel(const float* __restrict__ x,
                       const _Float16* __restrict__ W16,
                       float* __restrict__ out)
{
    __shared__ _Float16 lA[2][BM][LDSTRIDE];     // 20 KB
    __shared__ _Float16 lB[2][BN][LDSTRIDE];     // 40 KB

    const int tid  = threadIdx.x;
    const int lane = tid & 31;                   // wave32
    const int wid  = tid >> 5;
    const int wm   = wid & 1;                    // wave M index (0..1), 64 rows
    const int wn   = wid >> 1;                   // wave N index (0..3), 64 cols
    const int m0   = blockIdx.x * BM;
    const int n0   = blockIdx.y * BN;

    // A staging: thread owns 16 contiguous fp32 of one A-tile row
    const int rowA = tid >> 1;                   // 0..127
    const int segA = (tid & 1) * 16;             // 0 or 16 (halves)
    const float* gA = x + (size_t)(m0 + rowA) * K_IN + segA;

    // B staging: thread owns one full B-tile row (32 halves = 4 x b128, async)
    const _Float16* gB = W16 + (size_t)(n0 + tid) * K_IN;
    const unsigned ldsB0 = lds_addr(&lB[0][tid][0]);
    const unsigned ldsB1 = lds_addr(&lB[1][tid][0]);

    v8f acc[4][4];
    {
        v8f z = {0.f, 0.f, 0.f, 0.f, 0.f, 0.f, 0.f, 0.f};
#pragma unroll
        for (int mi = 0; mi < 4; ++mi)
#pragma unroll
            for (int ni = 0; ni < 4; ++ni) acc[mi][ni] = z;
    }

    v4f ra[4];   // 16 fp32 of x in flight

    // ---- prologue: k-tile 0 -> LDS buffer 0 ----
#pragma unroll
    for (int c = 0; c < 4; ++c)
        async_b128(ldsB0 + c * 16, (const void*)(gB + c * 8));
#pragma unroll
    for (int i = 0; i < 4; ++i) ra[i] = *(const v4f*)(gA + i * 4);
    {
        v8h h0, h1;
#pragma unroll
        for (int i = 0; i < 8; ++i) {
            h0[i] = (_Float16)ra[i >> 2][i & 3];
            h1[i] = (_Float16)ra[2 + (i >> 2)][i & 3];
        }
        *(v8h*)&lA[0][rowA][segA]     = h0;
        *(v8h*)&lA[0][rowA][segA + 8] = h1;
    }
    wait_async0();
    __syncthreads();

    const int laneLo = lane & 15;                // row/col within 16
    const int laneHi = lane >> 4;                // K-half select

    for (int kt = 0; kt < K_TILES; ++kt) {
        const int cur = kt & 1;

        // ---- prefetch next k-tile (async B directly to LDS, A via regs) ----
        if (kt + 1 < K_TILES) {
            const _Float16* gB2 = gB + (kt + 1) * BK;
            const unsigned  ldsBn = cur ? ldsB0 : ldsB1;
#pragma unroll
            for (int c = 0; c < 4; ++c)
                async_b128(ldsBn + c * 16, (const void*)(gB2 + c * 8));
            const float* gA2 = gA + (kt + 1) * BK;
#pragma unroll
            for (int i = 0; i < 4; ++i) ra[i] = *(const v4f*)(gA2 + i * 4);
        }

        // ---- fragment loads (ISA 16-bit 16x32 A layout; B = W rows) ----
        v16h af[4], bf[4];
#pragma unroll
        for (int mi = 0; mi < 4; ++mi) {
            const _Float16* p = &lA[cur][wm * 64 + mi * 16 + laneLo][laneHi * 8];
            v8h lo = *(const v8h*)p;
            v8h hi = *(const v8h*)(p + 16);
#pragma unroll
            for (int i = 0; i < 8; ++i) { af[mi][i] = lo[i]; af[mi][i + 8] = hi[i]; }
        }
#pragma unroll
        for (int ni = 0; ni < 4; ++ni) {
            const _Float16* p = &lB[cur][wn * 64 + ni * 16 + laneLo][laneHi * 8];
            v8h lo = *(const v8h*)p;
            v8h hi = *(const v8h*)(p + 16);
#pragma unroll
            for (int i = 0; i < 8; ++i) { bf[ni][i] = lo[i]; bf[ni][i + 8] = hi[i]; }
        }

        // ---- 16 WMMAs ----
#pragma unroll
        for (int mi = 0; mi < 4; ++mi)
#pragma unroll
            for (int ni = 0; ni < 4; ++ni)
                acc[mi][ni] = __builtin_amdgcn_wmma_f32_16x16x32_f16(
                    false, af[mi], false, bf[ni],
                    (short)0, acc[mi][ni], false, false);

        // ---- commit prefetched A, wait async B, flip buffers ----
        if (kt + 1 < K_TILES) {
            const int nxt = cur ^ 1;
            v8h h0, h1;
#pragma unroll
            for (int i = 0; i < 8; ++i) {
                h0[i] = (_Float16)ra[i >> 2][i & 3];
                h1[i] = (_Float16)ra[2 + (i >> 2)][i & 3];
            }
            *(v8h*)&lA[nxt][rowA][segA]     = h0;
            *(v8h*)&lA[nxt][rowA][segA + 8] = h1;
            wait_async0();
            __syncthreads();
        }
    }

    // ---- epilogue: VGPR r holds M = r + 8*(lane>=16), N = lane&15 ----
#pragma unroll
    for (int mi = 0; mi < 4; ++mi) {
#pragma unroll
        for (int ni = 0; ni < 4; ++ni) {
            const int col = n0 + wn * 64 + ni * 16 + laneLo;
#pragma unroll
            for (int r = 0; r < 8; ++r) {
                const int row = m0 + wm * 64 + mi * 16 + laneHi * 8 + r;
                out[(size_t)row * N_OUT + col] = acc[mi][ni][r];
            }
        }
    }
}

// ---------------------------------------------------------------------------
extern "C" void kernel_launch(void* const* d_in, const int* in_sizes, int n_in,
                              void* d_out, int out_size, void* d_ws, size_t ws_size,
                              hipStream_t stream)
{
    (void)in_sizes; (void)n_in; (void)out_size; (void)ws_size;
    const float*    x      = (const float*)d_in[0];
    const int*      bp     = (const int*)d_in[1];      // [OUT, IN/2] packed nibbles
    const float*    scales = (const float*)d_in[2];    // [OUT]
    const _Float16* vals   = (const _Float16*)d_in[3]; // [NNZ] fp16
    const int*      cols   = (const int*)d_in[4];      // [NNZ]
    // d_in[5] = crow ptr (fixed NNZ_PER_ROW per row, unused)

    _Float16* W16 = (_Float16*)d_ws;                   // 33.5 MB scratch
    float*    out = (float*)d_out;

    ortho_dequant_kernel<<<N_OUT, 256, 0, stream>>>(bp, scales, vals, cols, W16);

    dim3 grid(M_ROWS / BM, N_OUT / BN);                // 64 x 16
    ortho_gemm_kernel<<<grid, 256, 0, stream>>>(x, W16, out);
}
